// CIN_54065048322688
// MI455X (gfx1250) — compile-verified, matrix operand order checked by reading, not run on hardware
//
#include <hip/hip_runtime.h>
#include <hip/hip_bf16.h>

typedef __attribute__((ext_vector_type(16))) __bf16 v16bf;
typedef __attribute__((ext_vector_type(8)))  float  v8f;

union BFrag { uint4 u[2]; v16bf v; };
union BPack { __bf16 h[8]; uint4 u; };

static __device__ __forceinline__ v16bf splat16(__bf16 s) {
    v16bf v;
#pragma unroll
    for (int i = 0; i < 16; ++i) v[i] = s;
    return v;
}

// ---------------------------------------------------------------------------
// Zero-init pooled output (harness poisons d_out; we accumulate with atomics).
// ---------------------------------------------------------------------------
__global__ void zero_f32(float* __restrict__ p, int n) {
    int i = blockIdx.x * blockDim.x + threadIdx.x;
    if (i < n) p[i] = 0.0f;
}

// ---------------------------------------------------------------------------
// Repack W [128, 39*M] f32 row-major  ->  Wp [39][128][MPAD] bf16, zero padded.
// ---------------------------------------------------------------------------
__global__ void pack_w(const float* __restrict__ W, __bf16* __restrict__ Wp,
                       int M, int MPAD) {
    int idx = blockIdx.x * blockDim.x + threadIdx.x;
    int total = 39 * 128 * MPAD;
    if (idx >= total) return;
    int m = idx % MPAD;
    int t = idx / MPAD;
    int o = t % 128;
    int h = t / 128;
    float v = (m < M) ? W[(size_t)o * (39 * M) + h * M + m] : 0.0f;
    Wp[idx] = (__bf16)v;
}

// ---------------------------------------------------------------------------
// Fused CIN layer.
//   out[b,o,d] = sum_h X0[b,h,d] * ( sum_m Wp[h,o,m] * Xi[b,m,d] ) + bias[o]
// One workgroup per batch element b; 8 wave32s; wave owns a 32x32 (o,d) tile.
// B operand built on the fly: XiT (LDS, bf16) scaled by X0T[d][h] (bf16,
// broadcast via v_pk_mul_bf16 op_sel). WMMA: v_wmma_f32_16x16x32_bf16.
// Epilogue: +bias, shfl-reduce over d -> atomicAdd pooled; store Xnext as
// transposed bf16 [b][d][o] so the next layer stages LDS with async copies.
// ---------------------------------------------------------------------------
template<int MPAD, bool FIRST, bool STORE_NEXT>
__global__ __launch_bounds__(256) void cin_layer(
    const float* __restrict__ X0,       // [B,39,64] f32
    const __bf16* __restrict__ XiT_g,   // [B,64,128] bf16 (ignored when FIRST)
    const __bf16* __restrict__ Wp,      // [39,128,MPAD] bf16
    const float* __restrict__ bias,     // [128] f32
    float* __restrict__ out_pooled,     // d_out + layer*128 (row stride 384)
    __bf16* __restrict__ XnT_g)         // [B,64,128] bf16 (when STORE_NEXT)
{
    constexpr int XPITCH = MPAD + 8;            // stride mult of 16B, banks ok
    __shared__ __align__(16) __bf16 XiT[64][XPITCH];
    __shared__ __align__(16) __bf16 X0T[64][40];

    const int b   = blockIdx.x;
    const int tid = threadIdx.x;

    // ---- stage Xi^T (GEMM B source) ----
    if constexpr (!FIRST) {
        // Async VMEM -> LDS copy (no VGPR round trip), tracked by ASYNCcnt.
        const __bf16* src = XiT_g + (size_t)b * 64 * 128;
        for (int i = tid; i < 64 * 16; i += 256) {   // 16 x b128 per row
            int d = i >> 4, c = i & 15;
            unsigned lds = (unsigned)(size_t)&XiT[d][c * 8];
            unsigned long long ga = (unsigned long long)(size_t)(src + i * 8);
            asm volatile("global_load_async_to_lds_b128 %0, %1, off"
                         :: "v"(lds), "v"(ga) : "memory");
        }
    }
    // ---- stage X0^T (scales) ----
    for (int i = tid; i < 39 * 64; i += 256) {
        int h = i >> 6, d = i & 63;
        X0T[d][h] = (__bf16)X0[(size_t)b * (39 * 64) + i];
    }
    if constexpr (FIRST) {
        // Xi == X0 : transpose + cvt + zero-pad rows h >= 39
        for (int i = tid; i < 64 * 64; i += 256) {
            int h = i >> 6, d = i & 63;
            float v = (h < 39) ? X0[(size_t)b * (39 * 64) + i] : 0.0f;
            XiT[d][h] = (__bf16)v;
        }
    } else {
        asm volatile("s_wait_asynccnt 0" ::: "memory");
    }
    __syncthreads();

    const int wave = tid >> 5;
    const int lane = tid & 31;
    const int lhi  = lane >> 4;     // which K-half of the fragment this lane holds
    const int llo  = lane & 15;     // row (A: o, B: d) within tile
    const int ob   = (wave & 3) * 32;
    const int db   = (wave >> 2) * 32;

    v8f acc[2][2] = {};

    constexpr int KCH = MPAD / 32;
    for (int h = 0; h < 39; ++h) {
        // warm WGP$ for the next h-block of packed weights
        if (h + 1 < 39) {
            const __bf16* nw = Wp + (size_t)((h + 1) * 128 + ob + llo) * MPAD + lhi * 8;
            __builtin_prefetch(nw, 0, 0);
            __builtin_prefetch(nw + (size_t)16 * MPAD, 0, 0);
        }

        const __bf16 s0 = X0T[db + llo][h];
        const __bf16 s1 = X0T[db + 16 + llo][h];
        const v16bf sv0 = splat16(s0);
        const v16bf sv1 = splat16(s1);
#pragma unroll
        for (int kk = 0; kk < KCH; ++kk) {
            const int kofs = kk * 32 + lhi * 8;

            // A fragments: Wp rows (global, L2-resident, shared by all blocks)
            const __bf16* wr0 = Wp + (size_t)(h * 128 + ob + llo) * MPAD + kofs;
            const __bf16* wr1 = wr0 + (size_t)16 * MPAD;
            BFrag A0, A1, B0, B1;
            A0.u[0] = *(const uint4*)(wr0);
            A0.u[1] = *(const uint4*)(wr0 + 16);
            A1.u[0] = *(const uint4*)(wr1);
            A1.u[1] = *(const uint4*)(wr1 + 16);

            // B fragments: XiT rows (LDS), then scale by X0[b,h,d]
            const __bf16* xr0 = &XiT[db + llo][kofs];
            const __bf16* xr1 = &XiT[db + 16 + llo][kofs];
            B0.u[0] = *(const uint4*)(xr0);
            B0.u[1] = *(const uint4*)(xr0 + 16);
            B1.u[0] = *(const uint4*)(xr1);
            B1.u[1] = *(const uint4*)(xr1 + 16);

            const v16bf b0 = B0.v * sv0;      // -> v_pk_mul_bf16 (op_sel bcast)
            const v16bf b1 = B1.v * sv1;

            acc[0][0] = __builtin_amdgcn_wmma_f32_16x16x32_bf16(
                false, A0.v, false, b0, (short)0, acc[0][0], false, false);
            acc[0][1] = __builtin_amdgcn_wmma_f32_16x16x32_bf16(
                false, A0.v, false, b1, (short)0, acc[0][1], false, false);
            acc[1][0] = __builtin_amdgcn_wmma_f32_16x16x32_bf16(
                false, A1.v, false, b0, (short)0, acc[1][0], false, false);
            acc[1][1] = __builtin_amdgcn_wmma_f32_16x16x32_bf16(
                false, A1.v, false, b1, (short)0, acc[1][1], false, false);
        }
    }

    // ---- epilogue: bias, pooled reduction, transposed store ----
    float bv[2][8];
#pragma unroll
    for (int i = 0; i < 2; ++i) {
        const float4* bp = (const float4*)(bias + ob + i * 16 + lhi * 8);
        float4 u0 = bp[0], u1 = bp[1];
        bv[i][0] = u0.x; bv[i][1] = u0.y; bv[i][2] = u0.z; bv[i][3] = u0.w;
        bv[i][4] = u1.x; bv[i][5] = u1.y; bv[i][6] = u1.z; bv[i][7] = u1.w;
    }

#pragma unroll
    for (int i = 0; i < 2; ++i)
#pragma unroll
        for (int j = 0; j < 2; ++j)
#pragma unroll
            for (int r = 0; r < 8; ++r)
                acc[i][j][r] += bv[i][r];

    // pooled[b, o] = sum_d out[b,o,d]  (reduce across the 16 lanes of a tile row)
#pragma unroll
    for (int i = 0; i < 2; ++i)
#pragma unroll
        for (int j = 0; j < 2; ++j)
#pragma unroll
            for (int r = 0; r < 8; ++r) {
                float v = acc[i][j][r];
                v += __shfl_xor(v, 1, 32);
                v += __shfl_xor(v, 2, 32);
                v += __shfl_xor(v, 4, 32);
                v += __shfl_xor(v, 8, 32);
                if (llo == 0) {
                    int o = ob + i * 16 + lhi * 8 + r;
                    atomicAdd(&out_pooled[(size_t)b * 384 + o], v);
                }
            }

    if constexpr (STORE_NEXT) {
#pragma unroll
        for (int i = 0; i < 2; ++i)
#pragma unroll
            for (int j = 0; j < 2; ++j) {
                BPack pk;
#pragma unroll
                for (int r = 0; r < 8; ++r) pk.h[r] = (__bf16)acc[i][j][r];
                const int d = db + j * 16 + llo;
                const int o = ob + i * 16 + lhi * 8;
                *(uint4*)(XnT_g + ((size_t)b * 64 + d) * 128 + o) = pk.u;
            }
    }
}

// ---------------------------------------------------------------------------
// Launch: pack weights -> zero out -> 3 fused layers (intermediates stay in
// workspace as transposed bf16, resident in the 192 MB L2).
// ---------------------------------------------------------------------------
extern "C" void kernel_launch(void* const* d_in, const int* in_sizes, int n_in,
                              void* d_out, int out_size, void* d_ws, size_t ws_size,
                              hipStream_t stream) {
    const float* X0 = (const float*)d_in[0];
    const float* W1 = (const float*)d_in[1];
    const float* b1 = (const float*)d_in[2];
    const float* W2 = (const float*)d_in[3];
    const float* b2 = (const float*)d_in[4];
    const float* W3 = (const float*)d_in[5];
    const float* b3 = (const float*)d_in[6];
    float* out = (float*)d_out;

    char* ws = (char*)d_ws;
    __bf16* W1p = (__bf16*)(ws + 0x000000);                 // 39*128* 64*2 = 624 KB
    __bf16* W2p = (__bf16*)(ws + 0x100000);                 // 39*128*128*2 = 1.22 MB
    __bf16* W3p = (__bf16*)(ws + 0x280000);
    __bf16* X1T = (__bf16*)(ws + 0x400000);                 // 1024*64*128*2 = 16 MB
    __bf16* X2T = (__bf16*)(ws + 0x400000 + (size_t)(16u << 20));

    // repack weights to bf16 [h][o][m_pad]
    {
        int n1 = 39 * 128 * 64;
        pack_w<<<(n1 + 255) / 256, 256, 0, stream>>>(W1, W1p, 39, 64);
        int n2 = 39 * 128 * 128;
        pack_w<<<(n2 + 255) / 256, 256, 0, stream>>>(W2, W2p, 128, 128);
        pack_w<<<(n2 + 255) / 256, 256, 0, stream>>>(W3, W3p, 128, 128);
    }
    // zero pooled output (accumulated via atomics)
    {
        int n = 1024 * 384;
        zero_f32<<<(n + 255) / 256, 256, 0, stream>>>(out, n);
    }

    // layer 1: Xi = X0 (M=39 padded to 64)
    cin_layer<64, true, true><<<1024, 256, 0, stream>>>(
        X0, nullptr, W1p, b1, out + 0, X1T);
    // layer 2: Xi = X1 (M=128)
    cin_layer<128, false, true><<<1024, 256, 0, stream>>>(
        X0, X1T, W2p, b2, out + 128, X2T);
    // layer 3: Xi = X2 (M=128), no next-layer store
    cin_layer<128, false, false><<<1024, 256, 0, stream>>>(
        X0, X2T, W3p, b3, out + 256, nullptr);
}
